// LambdaMixingAttention_37864431681999
// MI455X (gfx1250) — matrix operand order repaired and checked by reading, not compiled, not run on hardware
//
#include <hip/hip_runtime.h>
#include <math.h>

// ---------------------------------------------------------------------------
// LambdaMixingAttention forward for MI455X (gfx1250, wave32, WMMA bf16)
//
// Shapes: B_=1024 windows, N=49 tokens, dim=256, d=128, KD=16, NH=8,
//         B=16 images, H=W=56, HW=3136, M_TOK = 50176 tokens.
//
// Heavy GEMMs run as bf16 WMMA (v_wmma_f32_16x16x32_bf16) with f32 accum.
// Activations/weights are converted to bf16 ONCE (or produced directly in
// bf16 by the upstream kernel), so GEMM inner loops are pure b128 loads +
// WMMA (no per-iteration cvt VALU). Each wave computes a 32x32 tile: 2 A-
// frags x 2 B-frags -> 4 WMMAs per k-step, 2 loads per WMMA.
// position_lambdas is factored: pos_out = (q.E) @ v -> avoids 411MB tensor.
// ---------------------------------------------------------------------------

#define EPSF 1e-5f
typedef unsigned short u16;

typedef __attribute__((ext_vector_type(16))) __bf16 v16bf;
typedef __attribute__((ext_vector_type(8)))  float  v8f;

static __device__ __forceinline__ u16 f2bf_bits(float f) {
  unsigned u = __builtin_bit_cast(unsigned, f);
  u += 0x7fffu + ((u >> 16) & 1u);              // round-to-nearest-even
  return (u16)(u >> 16);
}

static __device__ __forceinline__ float geluf(float x) {
  return 0.5f * x * (1.0f + erff(x * 0.70710678118654752440f));
}

static __device__ __forceinline__ float blkReduce(float v, float* red) {
  int t = threadIdx.x;
  red[t] = v; __syncthreads();
  for (int s = blockDim.x >> 1; s > 0; s >>= 1) {
    if (t < s) red[t] += red[t + s];
    __syncthreads();
  }
  float r = red[0]; __syncthreads();
  return r;
}

// A-fragment (16x32 bf16): lane m=l%16, hi=l/16.
// VGPR j: K = (j/4)*16 + hi*8 + (j%4)*2 -> two contiguous 8-elem runs:
//   [k0+hi*8, +8) and [k0+16+hi*8, +8)  == two 16B loads.
static __device__ __forceinline__ v16bf ldfragA(const u16* __restrict__ row,
                                                int k0, int hi) {
  struct P { uint4 x, y; } t;
  t.x = *(const uint4*)(row + k0 + hi * 8);
  t.y = *(const uint4*)(row + k0 + 16 + hi * 8);
  return __builtin_bit_cast(v16bf, t);
}

// B-fragment (32x16 bf16): lane n=l%16, hi=l/16; same K order as A-fragment:
//   [k0+hi*8, +8) and [k0+16+hi*8, +8)  == two 16B loads.
static __device__ __forceinline__ v16bf ldfragB(const u16* __restrict__ row,
                                                int k0, int hi) {
  return ldfragA(row, k0, hi);
}

// ---------------------------------------------------------------------------
// bf16 GEMM: out[m,n] = sum_k A[m,k]*W[n,k] + bias[n]
// A: (M,K) bf16 row-major, W: (N,K) bf16 row-major.
// Block: 256 thr = 8 waves (4m x 2n), block tile 128x64; wave tile 32x32.
// Requires M%128==0, N%64==0, K%32==0 (all call sites satisfy).
// D tile (16x16 f32): VGPR r -> row r + hi*8, col l%16.
// ---------------------------------------------------------------------------
__global__ void __launch_bounds__(256) gemm_bf16_wmma(
    const u16* __restrict__ A, const u16* __restrict__ W,
    const float* __restrict__ bias, float* __restrict__ out,
    int M, int N, int K)
{
  const int lane = threadIdx.x & 31;
  const int wave = threadIdx.x >> 5;
  const int nbn  = N >> 6;
  const int bm   = blockIdx.x / nbn;
  const int bn   = blockIdx.x % nbn;
  const int lr = lane & 15, hi = lane >> 4;
  const int m0 = bm * 128 + (wave >> 1) * 32;
  const int n0 = bn * 64 + (wave & 1) * 32;
  const u16* a0p = A + (size_t)(m0 + lr) * K;
  const u16* a1p = A + (size_t)(m0 + 16 + lr) * K;
  const u16* b0p = W + (size_t)(n0 + lr) * K;
  const u16* b1p = W + (size_t)(n0 + 16 + lr) * K;
  v8f acc00 = {}, acc01 = {}, acc10 = {}, acc11 = {};
  for (int k0 = 0; k0 < K; k0 += 32) {
    if (k0 + 32 < K) {
      __builtin_prefetch(a0p + k0 + 32, 0, 1);
      __builtin_prefetch(a1p + k0 + 32, 0, 1);
    }
    v16bf a0 = ldfragA(a0p, k0, hi);
    v16bf a1 = ldfragA(a1p, k0, hi);
    v16bf b0 = ldfragB(b0p, k0, hi);
    v16bf b1 = ldfragB(b1p, k0, hi);
    acc00 = __builtin_amdgcn_wmma_f32_16x16x32_bf16(false, a0, false, b0, (short)0, acc00, false, false);
    acc01 = __builtin_amdgcn_wmma_f32_16x16x32_bf16(false, a0, false, b1, (short)0, acc01, false, false);
    acc10 = __builtin_amdgcn_wmma_f32_16x16x32_bf16(false, a1, false, b0, (short)0, acc10, false, false);
    acc11 = __builtin_amdgcn_wmma_f32_16x16x32_bf16(false, a1, false, b1, (short)0, acc11, false, false);
  }
  const float bv0 = bias ? bias[n0 + lr] : 0.0f;
  const float bv1 = bias ? bias[n0 + 16 + lr] : 0.0f;
#pragma unroll
  for (int r = 0; r < 8; ++r) {
    const int mr0 = m0 + hi * 8 + r;
    const int mr1 = mr0 + 16;
    out[(size_t)mr0 * N + n0 + lr]      = acc00[r] + bv0;
    out[(size_t)mr0 * N + n0 + 16 + lr] = acc01[r] + bv1;
    out[(size_t)mr1 * N + n0 + lr]      = acc10[r] + bv0;
    out[(size_t)mr1 * N + n0 + 16 + lr] = acc11[r] + bv1;
  }
}

// keys = x_atten @ w_k^T ; q = (x_atten @ w_q^T) * 0.25   (bf16 in, K=128)
__global__ void __launch_bounds__(256) gemm_kq_wmma(
    const u16* __restrict__ xatt, const u16* __restrict__ wk,
    const u16* __restrict__ wq, float* __restrict__ keys,
    float* __restrict__ qout)
{
  const int lane = threadIdx.x & 31, wave = threadIdx.x >> 5;
  const int lr = lane & 15, hi = lane >> 4;
  const int m0 = blockIdx.x * 64 + (wave >> 1) * 16;
  const int sel = wave & 1;
  const u16* arow = xatt + (size_t)(m0 + lr) * 128;
  const u16* wrow = (sel ? wq : wk) + (size_t)lr * 128;
  v8f acc = {};
#pragma unroll
  for (int k0 = 0; k0 < 128; k0 += 32) {
    v16bf a = ldfragA(arow, k0, hi);
    v16bf b = ldfragB(wrow, k0, hi);
    acc = __builtin_amdgcn_wmma_f32_16x16x32_bf16(false, a, false, b, (short)0, acc, false, false);
  }
  float* out = sel ? qout : keys;
  const float mul = sel ? 0.25f : 1.0f;   // q scale = head_dim^-0.5 = 16^-0.5
#pragma unroll
  for (int r = 0; r < 8; ++r)
    out[(size_t)(m0 + hi * 8 + r) * 16 + lr] = acc[r] * mul;
}

// proj2d 1x1 conv as GEMM: Y[b,o,p] = sum_c W[o,c]*X[b,c,p] + bias[o]
// A = W (128x256 f32 cvt inline, K-contig), B = X (f32, K stride = 3136).
__global__ void __launch_bounds__(256) gemm_proj2d_wmma(
    const float* __restrict__ W, const float* __restrict__ bias,
    const float* __restrict__ X, float* __restrict__ Y)
{
  const int lane = threadIdx.x & 31, wave = threadIdx.x >> 5;
  const int lr = lane & 15, hi = lane >> 4;
  const int b   = blockIdx.x / 196;          // 2 m-blocks * 98 n-blocks
  const int rem = blockIdx.x % 196;
  const int bm = rem / 98, bn = rem % 98;
  const int m0 = bm * 64 + (wave >> 1) * 16;
  const int n0 = bn * 32 + (wave & 1) * 16;
  const int m = m0 + lr, n = n0 + lr;
  const float* arow = W + (size_t)m * 256;
  const float* Xb = X + (size_t)b * 256 * 3136;
  v8f acc = {};
  for (int k0 = 0; k0 < 256; k0 += 32) {
    v16bf a, bfr;
#pragma unroll
    for (int j = 0; j < 8; ++j) {
      const int ka = k0 + ((j >> 2) << 4) + hi * 8 + ((j & 3) << 1);
      a[2 * j]     = __builtin_bit_cast(__bf16, f2bf_bits(arow[ka]));
      a[2 * j + 1] = __builtin_bit_cast(__bf16, f2bf_bits(arow[ka + 1]));
      const int kb = k0 + ((j >> 2) << 4) + hi * 8 + ((j & 3) << 1);
      bfr[2 * j]     = __builtin_bit_cast(__bf16, f2bf_bits(Xb[(size_t)kb * 3136 + n]));
      bfr[2 * j + 1] = __builtin_bit_cast(__bf16, f2bf_bits(Xb[(size_t)(kb + 1) * 3136 + n]));
    }
    acc = __builtin_amdgcn_wmma_f32_16x16x32_bf16(false, a, false, bfr, (short)0, acc, false, false);
  }
#pragma unroll
  for (int r = 0; r < 8; ++r) {
    const int row = m0 + hi * 8 + r;
    Y[(size_t)b * 128 * 3136 + (size_t)row * 3136 + n] = acc[r] + bias[row];
  }
}

// ---------------------------------------------------------------------------
// Elementwise / normalization kernels
// ---------------------------------------------------------------------------
__global__ void zero_f32(float* p, int n) {
  int i = blockIdx.x * blockDim.x + threadIdx.x;
  if (i < n) p[i] = 0.0f;
}

__global__ void cvt_bf16_k(const float* __restrict__ in, u16* __restrict__ out,
                           long n) {
  for (size_t i = (size_t)blockIdx.x * 256 + threadIdx.x; i < (size_t)n;
       i += (size_t)gridDim.x * 256)
    out[i] = f2bf_bits(in[i]);
}

// LayerNorm, L=128, one row per block; emits bf16 (GEMM operand)
__global__ void __launch_bounds__(128) ln128_bf16(const float* __restrict__ in,
    const float* __restrict__ g, const float* __restrict__ b,
    u16* __restrict__ out)
{
  __shared__ float red[128];
  size_t row = blockIdx.x;
  int t = threadIdx.x;
  float v = in[row * 128 + t];
  float mean = blkReduce(v, red) * (1.0f / 128.0f);
  float d = v - mean;
  float var = blkReduce(d * d, red) * (1.0f / 128.0f);
  float rs = rsqrtf(var + EPSF);
  out[row * 128 + t] = f2bf_bits(d * rs * g[t] + b[t]);
}

// LayerNorm L=256, scatter to image layout (window_reverse2, C=256)
__global__ void __launch_bounds__(256) ln256_img(const float* __restrict__ in,
    const float* __restrict__ g, const float* __restrict__ b,
    float* __restrict__ img)
{
  __shared__ float red[256];
  int tok = blockIdx.x, c = threadIdx.x;
  float v = in[(size_t)tok * 256 + c];
  float mean = blkReduce(v, red) * (1.0f / 256.0f);
  float d = v - mean;
  float var = blkReduce(d * d, red) * (1.0f / 256.0f);
  float rs = rsqrtf(var + EPSF);
  int b_ = tok / 49, n = tok % 49;
  int bi = b_ >> 6, win = b_ & 63;
  int h = (win >> 3) * 7 + n / 7, w = (win & 7) * 7 + n % 7;
  img[(((size_t)bi * 256 + c) * 56 + h) * 56 + w] = d * rs * g[c] + b[c];
}

// depthwise 3x3 'SAME' + bias; accumulates per-channel sum/sumsq for BN
__global__ void __launch_bounds__(256) dwconv3x3(const float* __restrict__ img,
    const float* __restrict__ wdw, const float* __restrict__ bdw,
    float* __restrict__ out, float* sum, float* sq)
{
  __shared__ float red[256];
  int bc = blockIdx.x, c = bc & 255;
  const float* src = img + (size_t)bc * 3136;
  float* dst = out + (size_t)bc * 3136;
  float k[9];
#pragma unroll
  for (int i = 0; i < 9; ++i) k[i] = wdw[c * 9 + i];
  float bias = bdw[c];
  float ls = 0.f, lq = 0.f;
  for (int p = threadIdx.x; p < 3136; p += 256) {
    int h = p / 56, w = p % 56;
    float acc = bias;
#pragma unroll
    for (int ky = 0; ky < 3; ++ky) {
      int y = h + ky - 1;
      if ((unsigned)y < 56u) {
#pragma unroll
        for (int kx = 0; kx < 3; ++kx) {
          int x = w + kx - 1;
          if ((unsigned)x < 56u) acc += src[y * 56 + x] * k[ky * 3 + kx];
        }
      }
    }
    dst[p] = acc; ls += acc; lq += acc * acc;
  }
  float ts = blkReduce(ls, red); if (!threadIdx.x) atomicAdd(&sum[c], ts);
  float tq = blkReduce(lq, red); if (!threadIdx.x) atomicAdd(&sq[c], tq);
}

// BN finalize: scale = g*rsqrt(var+eps), shift = b - mean*scale
__global__ void finalize_bn(const float* sum, const float* sq, float cnt,
                            const float* g, const float* b,
                            float* sc, float* sh, int n)
{
  int i = blockIdx.x * blockDim.x + threadIdx.x;
  if (i < n) {
    float m = sum[i] / cnt;
    float v = sq[i] / cnt - m * m;
    float s = g[i] * rsqrtf(v + EPSF);
    sc[i] = s; sh[i] = b[i] - m * s;
  }
}

// apply dw BN + exact GELU in place; emit pooled mean per (b,c)
__global__ void __launch_bounds__(256) dw_gelu_pool_k(float* __restrict__ x,
    const float* __restrict__ sc, const float* __restrict__ sh,
    float* __restrict__ pooled)
{
  __shared__ float red[256];
  int bc = blockIdx.x, c = bc & 255;
  float* p = x + (size_t)bc * 3136;
  float s = 0.f;
  float scv = sc[c], shv = sh[c];
  for (int i = threadIdx.x; i < 3136; i += 256) {
    float y = geluf(p[i] * scv + shv);
    p[i] = y; s += y;
  }
  float ts = blkReduce(s, red);
  if (!threadIdx.x) pooled[bc] = ts * (1.0f / 3136.0f);
}

// channel interaction: pooled(16,256) -> ci1(32)+BN(batch)+GELU -> ci2(128) -> sigmoid
__global__ void __launch_bounds__(256) ci_k(const float* __restrict__ pooled,
    const float* w1, const float* b1, const float* g, const float* bb,
    const float* w2, const float* b2, float* __restrict__ gate)
{
  __shared__ float sp[4096];
  __shared__ float t1[512];
  int t = threadIdx.x;
  for (int i = t; i < 4096; i += 256) sp[i] = pooled[i];
  __syncthreads();
  for (int o = t; o < 512; o += 256) {
    int b = o >> 5, j = o & 31;
    float acc = b1[j];
    const float* pb = sp + b * 256;
    const float* wj = w1 + j * 256;
    for (int c = 0; c < 256; ++c) acc += pb[c] * wj[c];
    t1[o] = acc;
  }
  __syncthreads();
  if (t < 32) {                       // BN over batch of 16 per channel j=t
    float sm = 0.f, s2 = 0.f;
    for (int b = 0; b < 16; ++b) { float x = t1[b * 32 + t]; sm += x; s2 += x * x; }
    float mean = sm / 16.f, var = s2 / 16.f - mean * mean;
    float sc = g[t] * rsqrtf(var + EPSF), sh = bb[t] - mean * sc;
    for (int b = 0; b < 16; ++b) t1[b * 32 + t] = geluf(t1[b * 32 + t] * sc + sh);
  }
  __syncthreads();
  for (int o = t; o < 2048; o += 256) {
    int b = o >> 7, oo = o & 127;
    float acc = b2[oo];
    const float* tb = t1 + b * 32;
    const float* wo = w2 + oo * 32;
#pragma unroll
    for (int j = 0; j < 32; ++j) acc += tb[j] * wo[j];
    gate[o] = 1.0f / (1.0f + expf(-acc));
  }
}

// softmax over the 49 positions, per (window, kdim)
__global__ void keys_softmax_k(float* __restrict__ keys) {
  int b_ = blockIdx.x, t = threadIdx.x;
  if (t < 16) {
    float* base = keys + (size_t)b_ * 784 + t;
    float mx = -3.0e38f;
    for (int m = 0; m < 49; ++m) mx = fmaxf(mx, base[m * 16]);
    float sm = 0.f;
    for (int m = 0; m < 49; ++m) { float e = expf(base[m * 16] - mx); base[m * 16] = e; sm += e; }
    float inv = 1.0f / sm;
    for (int m = 0; m < 49; ++m) base[m * 16] *= inv;
  }
}

// q BN stats: one block per kdim, reduce over 50176 tokens
__global__ void __launch_bounds__(256) qstats_k(const float* __restrict__ q,
                                                float* sum, float* sq)
{
  __shared__ float red[256];
  int kd = blockIdx.x;
  float s = 0.f, s2 = 0.f;
  for (int i = threadIdx.x; i < 50176; i += 256) {
    float v = q[(size_t)i * 16 + kd]; s += v; s2 += v * v;
  }
  float ts = blkReduce(s, red);  if (!threadIdx.x) sum[kd] = ts;
  float tq = blkReduce(s2, red); if (!threadIdx.x) sq[kd] = tq;
}

// gate v by sigmoid(ci) (per image, per channel); BN1d stats per position n
__global__ void __launch_bounds__(256) vgate_stats_k(float* __restrict__ v,
    const float* __restrict__ gate, float* sum, float* sq)
{
  __shared__ float red[256];
  int n = blockIdx.x;
  float ls = 0.f, lq = 0.f;
  for (int i = threadIdx.x; i < 131072; i += 256) {   // 1024 windows * 128 ch
    int b_ = i >> 7, c = i & 127;
    size_t a = ((size_t)b_ * 49 + n) * 128 + c;
    float val = v[a] * gate[(b_ >> 6) * 128 + c];
    v[a] = val; ls += val; lq += val * val;
  }
  float ts = blkReduce(ls, red); if (!threadIdx.x) sum[n] = ts;
  float tq = blkReduce(lq, red); if (!threadIdx.x) sq[n] = tq;
}

__global__ void vapply_k(float* __restrict__ v, const float* sc, const float* sh) {
  for (size_t i = (size_t)blockIdx.x * 256 + threadIdx.x; i < (size_t)50176 * 128;
       i += (size_t)gridDim.x * 256) {
    int n = (int)(i >> 7) % 49;
    v[i] = v[i] * sc[n] + sh[n];
  }
}

// content_lambda[b,k,c] = sum_m keys[b,m,k]*v[b,m,c]
__global__ void __launch_bounds__(256) content_lambda_k(
    const float* __restrict__ keys, const float* __restrict__ v,
    float* __restrict__ clam)
{
  __shared__ float sk[784];
  __shared__ float sv[6272];
  int b_ = blockIdx.x, t = threadIdx.x;
  for (int i = t; i < 784; i += 256)  sk[i] = keys[(size_t)b_ * 784 + i];
  for (int i = t; i < 6272; i += 256) sv[i] = v[(size_t)b_ * 6272 + i];
  __syncthreads();
  for (int o = t; o < 2048; o += 256) {
    int k = o >> 7, c = o & 127;
    float acc = 0.f;
    for (int m = 0; m < 49; ++m) acc += sk[m * 16 + k] * sv[m * 128 + c];
    clam[(size_t)b_ * 2048 + o] = acc;
  }
}

// s[b,n,m] = sum_k qn[b,n,k]*E[n,m,k]   (qn = BN-normalized q)
__global__ void __launch_bounds__(256) pos_scores_k(const float* __restrict__ q,
    const float* scq, const float* shq, const float* __restrict__ E,
    float* __restrict__ s)
{
  __shared__ float sq_[784];
  int b_ = blockIdx.x, t = threadIdx.x;
  for (int i = t; i < 784; i += 256) {
    int kd = i & 15;
    sq_[i] = q[(size_t)b_ * 784 + i] * scq[kd] + shq[kd];
  }
  __syncthreads();
  for (int o = t; o < 2401; o += 256) {
    int n = o / 49;
    const float* e = E + (size_t)o * 16;   // (n*49+m)*16
    const float* qq = sq_ + n * 16;
    float acc = 0.f;
#pragma unroll
    for (int k = 0; k < 16; ++k) acc += qq[k] * e[k];
    s[(size_t)b_ * 2401 + o] = acc;
  }
}

// content+position output, channel softmax, spatial scatter, LN(x_at) -> cat[:,0:128] (bf16)
// NOTE: reference reshape (B,1,v,n)->(B,49,128) is v-major: flat = v*49+n.
__global__ void __launch_bounds__(256) lambda_out_k(
    const float* __restrict__ q, const float* scq, const float* shq,
    const float* __restrict__ clam, const float* __restrict__ v,
    const float* __restrict__ s, const float* __restrict__ g_at,
    const float* __restrict__ b_at, float* __restrict__ spatial,
    u16* __restrict__ cat)
{
  __shared__ float sq_[784];
  __shared__ float sc_[2048];
  __shared__ float sv_[6272];
  __shared__ float so_[6272];
  int b_ = blockIdx.x, t = threadIdx.x;
  for (int i = t; i < 784; i += 256) {
    int kd = i & 15;
    sq_[i] = q[(size_t)b_ * 784 + i] * scq[kd] + shq[kd];
  }
  for (int i = t; i < 2048; i += 256) sc_[i] = clam[(size_t)b_ * 2048 + i];
  for (int i = t; i < 6272; i += 256) sv_[i] = v[(size_t)b_ * 6272 + i];
  __syncthreads();
  const float* srow = s + (size_t)b_ * 2401;
  for (int o = t; o < 6272; o += 256) {
    int vv = o / 49, n = o % 49;                 // flat = v*49+n (reshape order)
    const float* qq = sq_ + n * 16;
    float acc = 0.f;
#pragma unroll
    for (int k = 0; k < 16; ++k) acc += qq[k] * sc_[k * 128 + vv];
    const float* sr = srow + n * 49;
    for (int m = 0; m < 49; ++m) acc += sr[m] * sv_[m * 128 + vv];
    so_[o] = acc;
  }
  __syncthreads();
  if (t < 49) {                                   // softmax over 128 channels
    float* row = so_ + t * 128;
    float mx = -3.0e38f;
    for (int j = 0; j < 128; ++j) mx = fmaxf(mx, row[j]);
    float sm = 0.f;
    for (int j = 0; j < 128; ++j) { float e = expf(row[j] - mx); row[j] = e; sm += e; }
    float inv = 1.0f / sm;
    for (int j = 0; j < 128; ++j) row[j] *= inv;
  }
  __syncthreads();
  int bi = b_ >> 6, win = b_ & 63;
  int hb = (win >> 3) * 7, wb = (win & 7) * 7;
  for (int o = t; o < 6272; o += 256) {           // window_reverse2, C=128
    int n = o >> 7, c = o & 127;
    int h = hb + n / 7, w = wb + n % 7;
    spatial[(((size_t)bi * 128 + c) * 56 + h) * 56 + w] = so_[o];
  }
  if (t < 49) {                                   // x_at = LN(attn), bf16 out
    const float* row = so_ + t * 128;
    float sm = 0.f;
    for (int j = 0; j < 128; ++j) sm += row[j];
    float mean = sm / 128.f, vq = 0.f;
    for (int j = 0; j < 128; ++j) { float d = row[j] - mean; vq += d * d; }
    float rs = rsqrtf(vq / 128.f + EPSF);
    u16* dst = cat + ((size_t)b_ * 49 + t) * 256;
    for (int j = 0; j < 128; ++j)
      dst[j] = f2bf_bits((row[j] - mean) * rs * g_at[j] + b_at[j]);
  }
}

// si1: conv1x1 128->16 per pixel, with BN stats per output channel
__global__ void __launch_bounds__(256) si1_bnstats_k(
    const float* __restrict__ spatial, const float* __restrict__ w1,
    const float* __restrict__ b1, float* __restrict__ t1,
    float* sum, float* sq)
{
  __shared__ float sw[2048];
  __shared__ float red[256];
  for (int i = threadIdx.x; i < 2048; i += 256) sw[i] = w1[i];
  __syncthreads();
  int idx = blockIdx.x * 256 + threadIdx.x;       // exact: 196*256 = 50176
  int b = idx / 3136, p = idx % 3136;
  float acc[16];
#pragma unroll
  for (int j = 0; j < 16; ++j) acc[j] = b1[j];
  const float* xp = spatial + (size_t)b * 128 * 3136 + p;
  for (int c = 0; c < 128; ++c) {
    float x = xp[(size_t)c * 3136];
#pragma unroll
    for (int j = 0; j < 16; ++j) acc[j] += x * sw[j * 128 + c];
  }
  float* tp = t1 + (size_t)b * 16 * 3136 + p;
#pragma unroll
  for (int j = 0; j < 16; ++j) tp[(size_t)j * 3136] = acc[j];
  for (int j = 0; j < 16; ++j) {
    float ts = blkReduce(acc[j], red);          if (!threadIdx.x) atomicAdd(&sum[j], ts);
    float tq = blkReduce(acc[j] * acc[j], red); if (!threadIdx.x) atomicAdd(&sq[j], tq);
  }
}

// gate = sigmoid(conv1x1_16->1(gelu(bn(t1))))
__global__ void __launch_bounds__(256) si_gate_k(const float* __restrict__ t1,
    const float* sc, const float* sh, const float* w2, const float* b2,
    float* __restrict__ gate)
{
  int idx = blockIdx.x * 256 + threadIdx.x;
  int b = idx / 3136, p = idx % 3136;
  const float* tp = t1 + (size_t)b * 16 * 3136 + p;
  float acc = b2[0];
#pragma unroll
  for (int j = 0; j < 16; ++j)
    acc += w2[j] * geluf(tp[(size_t)j * 3136] * sc[j] + sh[j]);
  gate[idx] = 1.0f / (1.0f + expf(-acc));
}

// gated = gate * proj2d; BN stats per channel (128)
__global__ void __launch_bounds__(256) gate_mul_stats_k(
    const float* __restrict__ gate, const float* __restrict__ x,
    float* __restrict__ out, float* sum, float* sq)
{
  __shared__ float red[256];
  int bc = blockIdx.x, b = bc >> 7, c = bc & 127;
  const float* g = gate + (size_t)b * 3136;
  const float* xp = x + (size_t)bc * 3136;
  float* op = out + (size_t)bc * 3136;
  float ls = 0.f, lq = 0.f;
  for (int p = threadIdx.x; p < 3136; p += 256) {
    float v = g[p] * xp[p];
    op[p] = v; ls += v; lq += v * v;
  }
  float ts = blkReduce(ls, red); if (!threadIdx.x) atomicAdd(&sum[c], ts);
  float tq = blkReduce(lq, red); if (!threadIdx.x) atomicAdd(&sq[c], tq);
}

// apply conv BN + window_partition2 -> cat[:,128:256] (bf16)
__global__ void cat_scatter_k(const float* __restrict__ gated,
    const float* sc, const float* sh, u16* __restrict__ cat)
{
  for (size_t idx = (size_t)blockIdx.x * 256 + threadIdx.x;
       idx < (size_t)50176 * 128; idx += (size_t)gridDim.x * 256) {
    int tk = (int)(idx >> 7), c = (int)(idx & 127);
    int b_ = tk / 49, n = tk % 49;
    int b = b_ >> 6, win = b_ & 63;
    int h = (win >> 3) * 7 + n / 7, w = (win & 7) * 7 + n % 7;
    cat[(size_t)tk * 256 + 128 + c] = f2bf_bits(
        gated[(((size_t)b * 128 + c) * 3136) + h * 56 + w] * sc[c] + sh[c]);
  }
}

// ---------------------------------------------------------------------------
// Host launcher (all launches on `stream`, stream order carries dependencies)
// ---------------------------------------------------------------------------
extern "C" void kernel_launch(void* const* d_in, const int* in_sizes, int n_in,
                              void* d_out, int out_size, void* d_ws, size_t ws_size,
                              hipStream_t stream)
{
  (void)in_sizes; (void)n_in; (void)out_size; (void)ws_size;
  const float* x        = (const float*)d_in[0];
  const float* w_attn   = (const float*)d_in[1];
  const float* b_attn   = (const float*)d_in[2];
  const float* g_an     = (const float*)d_in[3];
  const float* b_an     = (const float*)d_in[4];
  const float* w_cnn    = (const float*)d_in[5];
  const float* b_cnn    = (const float*)d_in[6];
  const float* g_cn     = (const float*)d_in[7];
  const float* b_cn     = (const float*)d_in[8];
  const float* w_dw     = (const float*)d_in[9];
  const float* b_dw     = (const float*)d_in[10];
  const float* g_dwbn   = (const float*)d_in[11];
  const float* b_dwbn   = (const float*)d_in[12];
  const float* w_ci1    = (const float*)d_in[13];
  const float* b_ci1    = (const float*)d_in[14];
  const float* g_cibn   = (const float*)d_in[15];
  const float* b_cibn   = (const float*)d_in[16];
  const float* w_ci2    = (const float*)d_in[17];
  const float* b_ci2    = (const float*)d_in[18];
  const float* w_p2d    = (const float*)d_in[19];
  const float* b_p2d    = (const float*)d_in[20];
  const float* w_si1    = (const float*)d_in[21];
  const float* b_si1    = (const float*)d_in[22];
  const float* g_sibn   = (const float*)d_in[23];
  const float* b_sibn   = (const float*)d_in[24];
  const float* w_si2    = (const float*)d_in[25];
  const float* b_si2    = (const float*)d_in[26];
  const float* g_convn  = (const float*)d_in[27];
  const float* b_convn  = (const float*)d_in[28];
  const float* g_attn_n = (const float*)d_in[29];
  const float* b_attn_n = (const float*)d_in[30];
  const float* w_proj   = (const float*)d_in[31];
  const float* b_proj   = (const float*)d_in[32];
  const float* w_q      = (const float*)d_in[33];
  const float* w_k      = (const float*)d_in[34];
  const float* w_v      = (const float*)d_in[35];
  const float* g_bnv    = (const float*)d_in[36];
  const float* b_bnv    = (const float*)d_in[37];
  const float* g_bnq    = (const float*)d_in[38];
  const float* b_bnq    = (const float*)d_in[39];
  const float* E        = (const float*)d_in[40];
  float* out = (float*)d_out;

  // ---- workspace layout: f32 region, then bf16 (u16) region ----
  float* ws      = (float*)d_ws;
  float* bufA    = ws;                    // 6422528  : pre_att  -> spatial
  float* bufB    = bufA + 6422528;        // 12845056 : pre_cnn
  float* bufImg  = bufB + 12845056;       // 12845056 : x_cnn image
  float* bufDw   = bufImg + 12845056;     // 12845056 : dw conv  -> gated (6.4M)
  float* bufP2d  = bufDw + 12845056;      // 6422528  : proj2d
  float* bufV    = bufP2d + 6422528;      // 6422528  : v
  float* bufCl   = bufV + 6422528;        // 2097152  : content_lambda
  float* bufS    = bufCl + 2097152;       // 2458624  : q.E scores
  float* bufKeys = bufS + 2458624;        // 802816
  float* bufQ    = bufKeys + 802816;      // 802816
  float* bufT1   = bufQ + 802816;         // 802816   : si intermediate
  float* bufGate = bufT1 + 802816;        // 50176    : spatial gate
  float* bufPool = bufGate + 50176;       // 4096
  float* bufCi   = bufPool + 4096;        // 2048     : sigmoid(ci)
  float* st      = bufCi + 2048;          // 2048     : stats + scale/shift
  float* st_dw_s = st + 0;    float* st_dw_q = st + 256;
  float* sc_dw   = st + 512;  float* sh_dw   = st + 768;
  float* st_q_s  = st + 1024; float* st_q_q  = st + 1040;
  float* sc_q    = st + 1056; float* sh_q    = st + 1072;
  float* st_v_s  = st + 1088; float* st_v_q  = st + 1152;
  float* sc_v    = st + 1216; float* sh_v    = st + 1280;
  float* st_si_s = st + 1344; float* st_si_q = st + 1360;
  float* sc_si   = st + 1376; float* sh_si   = st + 1392;
  float* st_cv_s = st + 1408; float* st_cv_q = st + 1536;
  float* sc_cv   = st + 1664; float* sh_cv   = st + 1792;
  // bf16 region (u16 elements)
  u16* bf     = (u16*)(st + 2048);
  u16* xbf    = bf;                       // 12845056 : x in bf16
  u16* xattbf = xbf + 12845056;           // 6422528  : x_atten bf16
  u16* catbf  = xattbf + 6422528;         // 12845056 : [x_at | x_cnn] bf16
  u16* wabf   = catbf + 12845056;         // 32768 w_attn
  u16* wcbf   = wabf + 32768;             // 65536 w_cnn
  u16* wvbf   = wcbf + 65536;             // 16384 w_v
  u16* wpbf   = wvbf + 16384;             // 65536 w_proj
  u16* wkbf   = wpbf + 65536;             // 2048  w_k
  u16* wqbf   = wkbf + 2048;              // 2048  w_q

  zero_f32<<<8, 256, 0, stream>>>(st, 2048);

  // one-time bf16 conversions (GEMM operands)
  cvt_bf16_k<<<4096, 256, 0, stream>>>(x, xbf, 12845056L);
  cvt_bf16_k<<<32, 256, 0, stream>>>(w_attn, wabf, 32768L);
  cvt_bf16_k<<<64, 256, 0, stream>>>(w_cnn, wcbf, 65536L);
  cvt_bf16_k<<<16, 256, 0, stream>>>(w_v, wvbf, 16384L);
  cvt_bf16_k<<<64, 256, 0, stream>>>(w_proj, wpbf, 65536L);
  cvt_bf16_k<<<2, 256, 0, stream>>>(w_k, wkbf, 2048L);
  cvt_bf16_k<<<2, 256, 0, stream>>>(w_q, wqbf, 2048L);

  // input projections (WMMA bf16): 4 wmma per k-step, pure b128-load operands
  gemm_bf16_wmma<<<784, 256, 0, stream>>>(xbf, wabf, b_attn, bufA, 50176, 128, 256);
  gemm_bf16_wmma<<<1568, 256, 0, stream>>>(xbf, wcbf, b_cnn, bufB, 50176, 256, 256);
  ln128_bf16<<<50176, 128, 0, stream>>>(bufA, g_an, b_an, xattbf);
  ln256_img<<<50176, 256, 0, stream>>>(bufB, g_cn, b_cn, bufImg);

  // conv branch
  dwconv3x3<<<4096, 256, 0, stream>>>(bufImg, w_dw, b_dw, bufDw, st_dw_s, st_dw_q);
  finalize_bn<<<1, 256, 0, stream>>>(st_dw_s, st_dw_q, 50176.f, g_dwbn, b_dwbn, sc_dw, sh_dw, 256);
  dw_gelu_pool_k<<<4096, 256, 0, stream>>>(bufDw, sc_dw, sh_dw, bufPool);
  ci_k<<<1, 256, 0, stream>>>(bufPool, w_ci1, b_ci1, g_cibn, b_cibn, w_ci2, b_ci2, bufCi);
  gemm_proj2d_wmma<<<3136, 256, 0, stream>>>(w_p2d, b_p2d, bufDw, bufP2d);

  // lambda attention branch
  gemm_kq_wmma<<<784, 256, 0, stream>>>(xattbf, wkbf, wqbf, bufKeys, bufQ);
  keys_softmax_k<<<1024, 32, 0, stream>>>(bufKeys);
  qstats_k<<<16, 256, 0, stream>>>(bufQ, st_q_s, st_q_q);
  finalize_bn<<<1, 256, 0, stream>>>(st_q_s, st_q_q, 50176.f, g_bnq, b_bnq, sc_q, sh_q, 16);
  gemm_bf16_wmma<<<784, 256, 0, stream>>>(xattbf, wvbf, nullptr, bufV, 50176, 128, 128);
  vgate_stats_k<<<49, 256, 0, stream>>>(bufV, bufCi, st_v_s, st_v_q);
  finalize_bn<<<1, 256, 0, stream>>>(st_v_s, st_v_q, 131072.f, g_bnv, b_bnv, sc_v, sh_v, 49);
  vapply_k<<<4096, 256, 0, stream>>>(bufV, sc_v, sh_v);
  content_lambda_k<<<1024, 256, 0, stream>>>(bufKeys, bufV, bufCl);
  pos_scores_k<<<1024, 256, 0, stream>>>(bufQ, sc_q, sh_q, E, bufS);
  lambda_out_k<<<1024, 256, 0, stream>>>(bufQ, sc_q, sh_q, bufCl, bufV, bufS,
                                         g_attn_n, b_attn_n, bufA, catbf);

  // spatial interaction gating + conv BN + window partition
  si1_bnstats_k<<<196, 256, 0, stream>>>(bufA, w_si1, b_si1, bufT1, st_si_s, st_si_q);
  finalize_bn<<<1, 256, 0, stream>>>(st_si_s, st_si_q, 50176.f, g_sibn, b_sibn, sc_si, sh_si, 16);
  si_gate_k<<<196, 256, 0, stream>>>(bufT1, sc_si, sh_si, w_si2, b_si2, bufGate);
  gate_mul_stats_k<<<2048, 256, 0, stream>>>(bufGate, bufP2d, bufDw, st_cv_s, st_cv_q);
  finalize_bn<<<1, 256, 0, stream>>>(st_cv_s, st_cv_q, 50176.f, g_convn, b_convn, sc_cv, sh_cv, 128);
  cat_scatter_k<<<4096, 256, 0, stream>>>(bufDw, sc_cv, sh_cv, catbf);

  // final projection (WMMA bf16) -> d_out
  gemm_bf16_wmma<<<1568, 256, 0, stream>>>(catbf, wpbf, b_proj, out, 50176, 256, 256);
}
